// MRCGNN_77403900608998
// MI455X (gfx1250) — matrix operand order repaired and checked by reading, not compiled, not run on hardware
//
#include <hip/hip_runtime.h>

#define DD 128
#define BN_EPS 1e-5f

typedef __attribute__((ext_vector_type(16))) __bf16 v16bf;
typedef __attribute__((ext_vector_type(8)))  float  v8f;

union Frag16 { uint4 q[2]; v16bf v; };

__device__ __forceinline__ unsigned short f2bf(float f) {
  unsigned u = __float_as_uint(f);
  u += 0x7FFFu + ((u >> 16) & 1u);   // round-to-nearest-even
  return (unsigned short)(u >> 16);
}

// ---------------- utility / elementwise kernels ----------------

__global__ void k_zero_f32(float* __restrict__ p, int n) {
  int i = blockIdx.x * blockDim.x + threadIdx.x;
  if (i < n) p[i] = 0.f;
}

__global__ void k_cvt_bf16(const float* __restrict__ in, unsigned short* __restrict__ out, int n) {
  int i = blockIdx.x * blockDim.x + threadIdx.x;
  if (i < n) out[i] = f2bf(in[i]);
}

__global__ void k_deg_count(const int* __restrict__ dst, float* __restrict__ deg, int e) {
  int i = blockIdx.x * blockDim.x + threadIdx.x;
  if (i < e) atomicAdd(&deg[dst[i]], 1.0f);
}

__global__ void k_dinv(const float* __restrict__ deg, float* __restrict__ dinv, int n) {
  int i = blockIdx.x * blockDim.x + threadIdx.x;
  if (i < n) dinv[i] = rsqrtf(deg[i] + 1.0f);   // +1 = self loop; always > 0
}

// ---------------- WMMA GEMM: H[n][128] = Abf[n][128] @ W[128][128] ----------------
// 8 waves / block; each wave computes a 16-row strip x all 128 columns.
// W staged transposed (Wt[nOut][k]) in LDS as bf16 so B-fragments are contiguous.

__global__ __launch_bounds__(256)
void k_gemm_wmma(const unsigned short* __restrict__ Abf,   // bf16 activations [nrows][128]
                 const float* __restrict__ Wf,             // f32 weights W[k][nOut]
                 float* __restrict__ H, int nrows)
{
  __shared__ unsigned short Wt[DD * DD];   // 32KB, Wt[nOut*128 + k]
  const int tid = threadIdx.x;
  for (int idx = tid; idx < DD * DD; idx += 256) {
    int nOut = idx >> 7, k = idx & 127;
    Wt[idx] = f2bf(Wf[k * DD + nOut]);     // transpose on the fly (W stays L2-resident)
  }
  __syncthreads();

  const int wave = tid >> 5;
  const int lane = tid & 31;
  const int hi   = lane >> 4;              // lane half selects K sub-block (ISA A/B layout)
  const int m    = lane & 15;
  const int rowbase = (blockIdx.x * 8 + wave) * 16;

  int arow = rowbase + m;
  if (arow >= nrows) arow = nrows - 1;     // clamp (stores guarded) -> EXEC stays all-1s
  const unsigned short* ap = Abf + (size_t)arow * DD;

  v8f acc[8] = {};

  #pragma unroll
  for (int kb = 0; kb < DD; kb += 32) {
    Frag16 a;                               // A 16x32 bf16: elems 0-7 K=kb+hi*8.., 8-15 K=kb+16+hi*8..
    a.q[0] = *(const uint4*)(ap + kb + hi * 8);
    a.q[1] = *(const uint4*)(ap + kb + 16 + hi * 8);
    #pragma unroll
    for (int nt = 0; nt < 8; ++nt) {
      const unsigned short* bp = &Wt[(size_t)(nt * 16 + m) * DD + kb + hi * 8];
      Frag16 b;                             // B 32x16 bf16: elems 0-7 K=kb+hi*8.., 8-15 K=kb+16+hi*8.., N = m
      b.q[0] = *(const uint4*)(bp);
      b.q[1] = *(const uint4*)(bp + 16);
      acc[nt] = __builtin_amdgcn_wmma_f32_16x16x32_bf16(
          false, a.v, false, b.v, (short)0, acc[nt], false, false);
    }
  }

  #pragma unroll
  for (int nt = 0; nt < 8; ++nt) {
    #pragma unroll
    for (int r = 0; r < 8; ++r) {           // D layout: VGPR r -> row M = r + hi*8, col = m
      int row = rowbase + r + hi * 8;
      if (row < nrows)
        H[(size_t)row * DD + nt * 16 + m] = acc[nt][r];
    }
  }
}

// ---------------- graph aggregation ----------------

// agg[i] = h[i] * dinv[i]^2  (self-loop term; also initializes agg)
__global__ void k_selfloop(const float* __restrict__ H, const float* __restrict__ dinv,
                           float* __restrict__ agg, int total) {
  int i = blockIdx.x * blockDim.x + threadIdx.x;
  if (i < total) { float dv = dinv[i >> 7]; agg[i] = H[i] * dv * dv; }
}

// one wave per edge: 32 lanes x float4 = one 512B feature row; gathers/atomics hit L2
__global__ __launch_bounds__(256)
void k_edge_agg(const int* __restrict__ src, const int* __restrict__ dst,
                const float* __restrict__ dinv, const float* __restrict__ H,
                float* __restrict__ agg, int e)
{
  int eid  = blockIdx.x * 8 + (threadIdx.x >> 5);
  int lane = threadIdx.x & 31;
  if (eid >= e) return;
  int s = src[eid], d = dst[eid];
  float nrm = dinv[s] * dinv[d];
  float4 v = *(const float4*)(H + (size_t)s * DD + lane * 4);
  float* a = agg + (size_t)d * DD + lane * 4;
  atomicAdd(a + 0, v.x * nrm);
  atomicAdd(a + 1, v.y * nrm);
  atomicAdd(a + 2, v.z * nrm);
  atomicAdd(a + 3, v.w * nrm);
}

// ---------------- batch norm ----------------

__global__ void k_bn_partial(const float* __restrict__ agg,
                             float* __restrict__ colsum, float* __restrict__ colsq, int n) {
  int col = threadIdx.x;                    // 128 threads = coalesced row reads
  float s = 0.f, ss = 0.f;
  for (int r = blockIdx.x; r < n; r += gridDim.x) {
    float v = agg[(size_t)r * DD + col];
    s += v; ss += v * v;
  }
  atomicAdd(&colsum[col], s);
  atomicAdd(&colsq[col], ss);
}

__global__ void k_bn_final(const float* __restrict__ colsum, const float* __restrict__ colsq,
                           const float* __restrict__ gamma, const float* __restrict__ beta,
                           float* __restrict__ ss, int n) {
  int c = threadIdx.x;
  if (c < DD) {
    float inv_n = 1.0f / (float)n;
    float mean  = colsum[c] * inv_n;
    float var   = colsq[c] * inv_n - mean * mean;   // biased variance (training-mode BN)
    float sc    = gamma[c] * rsqrtf(var + BN_EPS);
    ss[c]       = sc;
    ss[DD + c]  = beta[c] - mean * sc;
  }
}

// out = relu(agg*scale + shift) + residual   (GCN bias cancels inside BN -> omitted)
__global__ void k_bn_relu_res(const float* __restrict__ agg, const float* __restrict__ ss,
                              const float* __restrict__ xin, float* __restrict__ out, int total) {
  int i = blockIdx.x * blockDim.x + threadIdx.x;
  if (i < total) {
    int c = i & 127;
    float v = fmaf(agg[i], ss[c], ss[DD + c]);
    out[i] = fmaxf(v, 0.f) + xin[i];
  }
}

// ---------------- attention (softmax over nodes) ----------------
// tanh in [-1,1] -> softmax is stable without max subtraction.

__global__ __launch_bounds__(256)
void k_att_logits(const float* __restrict__ H, const float* __restrict__ Wa,
                  const float* __restrict__ ba, float* __restrict__ att,
                  float* __restrict__ expsum, int n)
{
  int row  = blockIdx.x * 8 + (threadIdx.x >> 5);
  int lane = threadIdx.x & 31;
  if (row >= n) return;
  float4 h = *(const float4*)(H  + (size_t)row * DD + lane * 4);
  float4 w = *(const float4*)(Wa + lane * 4);
  float p = h.x * w.x + h.y * w.y + h.z * w.z + h.w * w.w;
  #pragma unroll
  for (int o = 16; o > 0; o >>= 1) p += __shfl_xor(p, o, 32);   // wave32 reduce
  if (lane == 0) {
    float e = __expf(tanhf(p + ba[0]));
    att[row] = e;
    atomicAdd(expsum, e);
  }
}

__global__ void k_att_apply(const float* __restrict__ H, const float* __restrict__ att,
                            const float* __restrict__ expsum, float* __restrict__ out, int total) {
  int i = blockIdx.x * blockDim.x + threadIdx.x;
  if (i < total) out[i] = H[i] * att[i >> 7] * (1.0f / expsum[0]);
}

// ---------------- driver ----------------

extern "C" void kernel_launch(void* const* d_in, const int* in_sizes, int n_in,
                              void* d_out, int out_size, void* d_ws, size_t ws_size,
                              hipStream_t stream) {
  (void)n_in; (void)out_size; (void)ws_size;
  const float* x   = (const float*)d_in[0];
  const int*   ei  = (const int*)d_in[1];
  const float* W1  = (const float*)d_in[2];
  const float* g1  = (const float*)d_in[4];
  const float* be1 = (const float*)d_in[5];
  const float* W2  = (const float*)d_in[6];
  const float* g2  = (const float*)d_in[8];
  const float* be2 = (const float*)d_in[9];
  const float* Wa  = (const float*)d_in[10];
  const float* ba  = (const float*)d_in[11];

  const int N  = in_sizes[0] / DD;
  const int E  = in_sizes[1] / 2;
  const int ND = N * DD;
  const int* src = ei;
  const int* dst = ei + E;

  char* ws = (char*)d_ws;
  size_t off = 0;
  auto alloc = [&](size_t bytes) -> char* {
    char* p = ws + off;
    off = (off + bytes + 255) & ~(size_t)255;
    return p;
  };
  float* A  = (float*)alloc((size_t)ND * 4);            // gemm tmp / layer1 out
  float* Bv = (float*)alloc((size_t)ND * 4);            // agg / layer2 out
  float* C  = (float*)alloc((size_t)ND * 4);            // layer2 gemm tmp
  unsigned short* XB = (unsigned short*)alloc((size_t)ND * 2);  // bf16 activations
  float* deg    = (float*)alloc((size_t)(N + 1) * 4);   // deg[N] + expsum
  float* expsum = deg + N;
  float* dinv   = (float*)alloc((size_t)N * 4);
  float* att    = (float*)alloc((size_t)N * 4);
  float* stats  = (float*)alloc(256 * 4);               // colsum[128] | colsq[128]
  float* ss     = (float*)alloc(256 * 4);               // scale[128] | shift[128]

  const int TB = 256;
  const int gND   = (ND + TB - 1) / TB;
  const int gN    = (N + TB - 1) / TB;
  const int gN1   = (N + 1 + TB - 1) / TB;
  const int gE    = (E + TB - 1) / TB;
  const int gGemm = (N + 127) / 128;      // 8 waves x 16 rows per block
  const int gWE   = (E + 7) / 8;          // wave per edge
  const int gWN   = (N + 7) / 8;          // wave per row

  // normalization terms (edge-structure only; shared by both layers) + expsum init
  k_zero_f32 <<<gN1, TB, 0, stream>>>(deg, N + 1);
  k_deg_count<<<gE,  TB, 0, stream>>>(dst, deg, E);
  k_dinv     <<<gN,  TB, 0, stream>>>(deg, dinv, N);

  // ---- layer 1 ----
  k_cvt_bf16   <<<gND,  TB, 0, stream>>>(x, XB, ND);
  k_gemm_wmma  <<<gGemm, TB, 0, stream>>>(XB, W1, A, N);
  k_selfloop   <<<gND,  TB, 0, stream>>>(A, dinv, Bv, ND);
  k_edge_agg   <<<gWE,  TB, 0, stream>>>(src, dst, dinv, A, Bv, E);
  k_zero_f32   <<<1,    TB, 0, stream>>>(stats, 256);
  k_bn_partial <<<256,  DD, 0, stream>>>(Bv, stats, stats + DD, N);
  k_bn_final   <<<1,    DD, 0, stream>>>(stats, stats + DD, g1, be1, ss, N);
  k_bn_relu_res<<<gND,  TB, 0, stream>>>(Bv, ss, x, A, ND);          // h1 -> A

  // ---- layer 2 ----
  k_cvt_bf16   <<<gND,  TB, 0, stream>>>(A, XB, ND);
  k_gemm_wmma  <<<gGemm, TB, 0, stream>>>(XB, W2, C, N);
  k_selfloop   <<<gND,  TB, 0, stream>>>(C, dinv, Bv, ND);
  k_edge_agg   <<<gWE,  TB, 0, stream>>>(src, dst, dinv, C, Bv, E);
  k_zero_f32   <<<1,    TB, 0, stream>>>(stats, 256);
  k_bn_partial <<<256,  DD, 0, stream>>>(Bv, stats, stats + DD, N);
  k_bn_final   <<<1,    DD, 0, stream>>>(stats, stats + DD, g2, be2, ss, N);
  k_bn_relu_res<<<gND,  TB, 0, stream>>>(Bv, ss, A, Bv, ND);         // h2 -> Bv (in-place ok)

  // ---- attention over nodes ----
  k_att_logits <<<gWN,  TB, 0, stream>>>(Bv, Wa, ba, att, expsum, N);
  k_att_apply  <<<gND,  TB, 0, stream>>>(Bv, att, expsum, (float*)d_out, ND);
}
